// CoPEWithFIRE_44641890074922
// MI455X (gfx1250) — matrix operand order repaired and checked by reading, not compiled, not run on hardware
//
#include <hip/hip_runtime.h>

typedef __attribute__((ext_vector_type(16))) _Float16 v16h;
typedef __attribute__((ext_vector_type(8)))  float    v8f;
typedef __attribute__((ext_vector_type(4)))  float    v4f;

#define SEQ   768
#define HEADS 12
#define WDIM  32
#define NWAVE (SEQ / 32)   // 24 waves per block

// One block per (head, query-row). 768 threads, one key-position j per thread.
__global__ __launch_bounds__(SEQ) void cope_fire_kernel(
    const float* __restrict__ logits,   // (H*S, S)
    const float* __restrict__ W_in,     // (32,1) -> w1[k]
    const float* __restrict__ b_in,     // (32,)
    const float* __restrict__ W_out,    // (12,32)
    const float* __restrict__ b_out,    // (12,)
    const float* __restrict__ c_p,      // scalar
    const float* __restrict__ Lm_p,     // scalar
    const float* __restrict__ iL_p,     // scalar
    float* __restrict__ out)            // (H*S, S)
{
    __shared__ float wsum[NWAVE];

    const int tid  = threadIdx.x;
    const int lane = tid & 31;
    const int wv   = tid >> 5;
    const int row  = blockIdx.x;        // row = h*768 + i
    const int h    = row / SEQ;

    const float c   = c_p[0];
    const float thr = fabsf(Lm_p[0] * iL_p[0]);   // |L_multiplier * init_L|

    // ---- per-lane MLP constants ------------------------------------------
    // A-matrix (16-bit, 16x32) lane layout: lanes 0-15 hold K in {0..7,16..23},
    // lanes 16-31 hold K in {8..15,24..31}, packed (VGPR v holds K pair).
    const int kbase = (lane >= 16) ? 8 : 0;
    float w1r[16], b1r[16];
    v16h Bm;
    #pragma unroll
    for (int i = 0; i < 16; ++i) {
        const int k = kbase + i + (i & 8);        // i<8 -> kbase+i ; i>=8 -> kbase+i+8
        w1r[i] = W_in[k];
        b1r[i] = b_in[k];
        // B (32x16): every column n is W_out[h]; same K order as A:
        // lanes 0-15 carry K in {0..7,16..23}, lanes 16-31 carry K in {8..15,24..31}.
        const int kb = kbase + i + (i & 8);
        Bm[i] = (_Float16)W_out[h * WDIM + kb];
    }

    // ---- 1. sigmoid -------------------------------------------------------
    const float x = logits[(size_t)row * SEQ + tid];
    const float g = 1.0f / (1.0f + __expf(-x));

    // ---- 2. reverse cumsum: pos[j] = T - exclusive_prefix[j] --------------
    float incl = g;
    #pragma unroll
    for (int d = 1; d < 32; d <<= 1) {
        float n = __shfl_up(incl, d);
        if (lane >= d) incl += n;
    }
    if (lane == 31) wsum[wv] = incl;
    __syncthreads();

    float T = 0.0f, off = 0.0f;
    #pragma unroll
    for (int w = 0; w < NWAVE; ++w) {
        const float s = wsum[w];
        T += s;
        off += (w < wv) ? s : 0.0f;
    }
    const float pos = T - (off + incl - g);

    // ---- 3. dist ----------------------------------------------------------
    const float num      = __logf(fabsf(c * pos) + 1.0f);
    const float den      = __logf(fabsf(c * fminf(T, thr)) + 1.0f) + 1e-6f;
    const float dist     = num / den;
    const float bout     = b_out[h];
    float* const orow    = out + (size_t)row * SEQ;
    const int    jw      = wv * 32;

    // ---- 4. MLP via WMMA: two 16-element tiles per wave -------------------
    #pragma unroll
    for (int t = 0; t < 2; ++t) {
        // A row m = lane%16 ; broadcast that element's dist to the lane.
        const float dm = __shfl(dist, t * 16 + (lane & 15));
        v16h Am;
        #pragma unroll
        for (int i = 0; i < 16; ++i) {
            float hv = fmaf(dm, w1r[i], b1r[i]);
            hv = fmaxf(hv, 0.0f);                 // relu
            Am[i] = (_Float16)hv;
        }
        v8f acc = {};
        acc = __builtin_amdgcn_wmma_f32_16x16x32_f16(
            /*neg_a=*/false, Am, /*neg_b=*/false, Bm,
            /*c_mod=*/(short)0, acc, /*reuse_a=*/false, /*reuse_b=*/false);

        // C/D layout: lane 0 holds D[M=0..7, N=0], lane 16 holds D[M=8..15, N=0];
        // all N columns are identical, so two lanes emit the 16 results.
        if ((lane & 15) == 0) {
            const int mbase = (lane >> 4) * 8;
            float* p = orow + jw + t * 16 + mbase;
            v4f lo = { acc[0] + bout, acc[1] + bout, acc[2] + bout, acc[3] + bout };
            v4f hi = { acc[4] + bout, acc[5] + bout, acc[6] + bout, acc[7] + bout };
            *(v4f*)(p)     = lo;
            *(v4f*)(p + 4) = hi;
        }
    }
}

extern "C" void kernel_launch(void* const* d_in, const int* in_sizes, int n_in,
                              void* d_out, int out_size, void* d_ws, size_t ws_size,
                              hipStream_t stream) {
    (void)in_sizes; (void)n_in; (void)out_size; (void)d_ws; (void)ws_size;
    const float* logits = (const float*)d_in[0];
    const float* W_in   = (const float*)d_in[1];
    const float* b_in   = (const float*)d_in[2];
    const float* W_out  = (const float*)d_in[3];
    const float* b_out  = (const float*)d_in[4];
    const float* c      = (const float*)d_in[5];
    const float* Lm     = (const float*)d_in[6];
    const float* iL     = (const float*)d_in[7];
    float* out = (float*)d_out;

    cope_fire_kernel<<<dim3(HEADS * SEQ), dim3(SEQ), 0, stream>>>(
        logits, W_in, b_in, W_out, b_out, c, Lm, iL, out);
}
